// QuantizedGroupEmbedding_26353919328819
// MI455X (gfx1250) — 8-bit matrix operand order repaired and checked by reading, compiled, not run on hardware
//
#include <hip/hip_runtime.h>
#include <hip/hip_fp16.h>


#define EMB    4096
#define GROUPS 16

typedef __attribute__((ext_vector_type(8))) int   v8i;
typedef __attribute__((ext_vector_type(2))) int   i2v;
typedef __attribute__((ext_vector_type(8))) float v8f;

__global__ __launch_bounds__(256) void
quant_group_embed_wmma_iu8(const signed char* __restrict__ weight,
                           const _Float16*    __restrict__ scales,
                           const int*         __restrict__ indices,
                           float*             __restrict__ out,
                           int num_tokens)
{
    const int tokBase = blockIdx.x * 16;
    if (tokBase >= num_tokens) return;

    const int tid   = threadIdx.x;
    const int lane  = tid & 31;
    const int wave  = tid >> 5;
    const int laneM = lane & 15;
    const int half  = lane >> 4;
    const int off   = half * 16;

    const int rI = indices[tokBase + laneM];
    const signed char* rowBase = weight + (size_t)rI * (size_t)EMB;

    const int colBase = wave * 512;

    const int unitByte = 1 << ((laneM & 3) * 8);
    const int q = laneM >> 2;
    v8i b0, b1, b2, b3;
#pragma unroll
    for (int e = 0; e < 8; ++e) {
        b0[e] = (half == 0 && e == q)     ? unitByte : 0;
        b1[e] = (half == 1 && e == q)     ? unitByte : 0;
        b2[e] = (half == 0 && e == q + 4) ? unitByte : 0;
        b3[e] = (half == 1 && e == q + 4) ? unitByte : 0;
    }

    const int mHi = half * 8;
    v8f srA, srB;
#pragma unroll
    for (int r = 0; r < 8; ++r) {
        const int ti = indices[tokBase + mHi + r];
        srA[r] = (float)scales[(size_t)ti * GROUPS + wave * 2 + 0];
        srB[r] = (float)scales[(size_t)ti * GROUPS + wave * 2 + 1];
    }

    auto loadA = [&](int n0) -> v8i {
        const i2v p0 = *(const i2v*)(rowBase + n0 + off);
        const i2v p1 = *(const i2v*)(rowBase + n0 + off + 8);
        const i2v p2 = *(const i2v*)(rowBase + n0 + off + 32);
        const i2v p3 = *(const i2v*)(rowBase + n0 + off + 40);
        v8i a;
        a[0] = p0[0]; a[1] = p0[1];
        a[2] = p1[0]; a[3] = p1[1];
        a[4] = p2[0]; a[5] = p2[1];
        a[6] = p3[0]; a[7] = p3[1];
        return a;
    };

    auto storePair = [&](int n0, int c0, const v8f& sv,
                         const v8i& dA, const v8i& dB) {
#pragma unroll
        for (int r = 0; r < 8; ++r) {
            const size_t row = (size_t)(tokBase + mHi + r) * EMB + n0 + laneM;
            __builtin_nontemporal_store(sv[r] * (float)dA[r], &out[row + c0]);
            __builtin_nontemporal_store(sv[r] * (float)dB[r], &out[row + c0 + 16]);
        }
    };

    const v8i z = {};

    v8i a  = loadA(colBase);
    v8i e0 = __builtin_amdgcn_wmma_i32_16x16x64_iu8(true, a, false, b0, z, false, false);
    v8i e1 = __builtin_amdgcn_wmma_i32_16x16x64_iu8(true, a, false, b1, z, false, false);
    v8i e2 = __builtin_amdgcn_wmma_i32_16x16x64_iu8(true, a, false, b2, z, false, false);
    v8i e3 = __builtin_amdgcn_wmma_i32_16x16x64_iu8(true, a, false, b3, z, false, false);

#pragma unroll 1
    for (int mt = 0; mt < 7; ++mt) {
        const int n0 = colBase + mt * 64;

        const v8i aN = loadA(n0 + 64);
        __builtin_amdgcn_sched_barrier(0);

        const v8f sv = (mt < 4) ? srA : srB;
        const v8i d0 = e0, d1 = e1, d2 = e2, d3 = e3;

        storePair(n0, 0, sv, d0, d1);
        __builtin_amdgcn_sched_barrier(0);

        e0 = __builtin_amdgcn_wmma_i32_16x16x64_iu8(true, aN, false, b0, z, false, false);
        e1 = __builtin_amdgcn_wmma_i32_16x16x64_iu8(true, aN, false, b1, z, false, false);
        __builtin_amdgcn_sched_barrier(0);

        storePair(n0, 32, sv, d2, d3);
        __builtin_amdgcn_sched_barrier(0);

        e2 = __builtin_amdgcn_wmma_i32_16x16x64_iu8(true, aN, false, b2, z, false, false);
        e3 = __builtin_amdgcn_wmma_i32_16x16x64_iu8(true, aN, false, b3, z, false, false);
        __builtin_amdgcn_sched_barrier(0);

        a = aN;
    }

    {
        const int n0 = colBase + 7 * 64;
        storePair(n0, 0,  srB, e0, e1);
        storePair(n0, 32, srB, e2, e3);
    }
}

extern "C" void kernel_launch(void* const* d_in, const int* in_sizes, int n_in,
                              void* d_out, int out_size, void* d_ws, size_t ws_size,
                              hipStream_t stream) {
    (void)d_ws; (void)ws_size; (void)out_size; (void)n_in;

    const signed char* weight  = (const signed char*)d_in[0];
    const _Float16*    scales  = (const _Float16*)   d_in[1];
    const int*         indices = (const int*)        d_in[2];
    float*             out     = (float*)d_out;

    const int num_tokens = in_sizes[2];
    const int blocks     = (num_tokens + 15) / 16;

    quant_group_embed_wmma_iu8<<<blocks, 256, 0, stream>>>(
        weight, scales, indices, out, num_tokens);
}
